// MultiHeadAttention_12764642804143
// MI455X (gfx1250) — matrix operand order repaired and checked by reading, not compiled, not run on hardware
//
#include <hip/hip_runtime.h>

#define B_ 2
#define S_ 2048
#define D_ 1024
#define H_ 16
#define DH_ 64
#define NEGV -1000000000.0f

typedef __attribute__((ext_vector_type(16))) __bf16 v16bf;
typedef __attribute__((ext_vector_type(8))) float v8f;
typedef __attribute__((ext_vector_type(4))) unsigned int u32x4;
typedef __attribute__((ext_vector_type(4))) int i32x4;
typedef __attribute__((ext_vector_type(8))) int i32x8;

#if defined(__has_builtin)
#if __has_builtin(__builtin_amdgcn_tensor_load_to_lds) && \
    __has_builtin(__builtin_amdgcn_s_wait_tensorcnt)
#define HAVE_TDM 1
#endif
#endif
#ifndef HAVE_TDM
#define HAVE_TDM 0
#endif

union FragB {
  v16bf f;
  u32x4 q[2];
};

union Pack8 {
  u32x4 v;
  unsigned short e[8];
};

__device__ __forceinline__ unsigned short f2bf(float x) {
  unsigned int u = __builtin_bit_cast(unsigned int, x);
  u += 0x7fffu + ((u >> 16) & 1u);  // round-to-nearest-even
  return (unsigned short)(u >> 16);
}

#if HAVE_TDM
// Issue a 2D TDM tile load (D# per CDNA5 ISA ch.8). All fields wave-uniform.
__device__ __forceinline__ void tdm_load_2d(
    unsigned lds_off, const void* gptr,
    unsigned tensor_d0, unsigned tensor_d1, unsigned tile_d0, unsigned tile_d1,
    unsigned stride0, unsigned pad_interval_log, unsigned pad_amount_dw) {
  unsigned long long ga = (unsigned long long)(uintptr_t)gptr;
  u32x4 g0;
  g0[0] = 1u;  // count=1 (valid), is_restore=0, gather off
  g0[1] = lds_off;
  g0[2] = (unsigned)(ga & 0xffffffffu);
  g0[3] = (unsigned)((ga >> 32) & 0x01ffffffu) | (2u << 30);  // type=2 (image)
  i32x8 g1;
  // data_size=1 (2B) | pad_enable | pad_interval | pad_amount
  g1[0] = (int)((1u << 16) | (1u << 20) | (pad_interval_log << 22) |
                (pad_amount_dw << 25));
  g1[1] = (int)((tensor_d0 & 0xffffu) << 16);                     // dim0 lo
  g1[2] = (int)(((tensor_d0 >> 16) & 0xffffu) |
                ((tensor_d1 & 0xffffu) << 16));                   // dim0 hi|dim1 lo
  g1[3] = (int)(((tensor_d1 >> 16) & 0xffffu) | (tile_d0 << 16)); // dim1 hi|tile0
  g1[4] = (int)(tile_d1 & 0xffffu);                               // tile1, tile2=0
  g1[5] = (int)stride0;                                           // dim0 stride lo
  g1[6] = 0;                                                      // stride hi, dim1 stride
  g1[7] = 0;
  i32x4 gz = {};
#if defined(__clang_major__) && __clang_major__ >= 23
  i32x8 gz8 = {};
  __builtin_amdgcn_tensor_load_to_lds(g0, g1, gz, gz, gz8, 0);
#else
  __builtin_amdgcn_tensor_load_to_lds(g0, g1, gz, gz, 0);
#endif
}
#endif

// ---------------------------------------------------------------------------
// fp32 -> bf16 conversion
// ---------------------------------------------------------------------------
__global__ void cvt_f32_to_bf16(const float* __restrict__ in,
                                unsigned short* __restrict__ out, int n) {
  int i = blockIdx.x * blockDim.x + threadIdx.x;
  if (i < n) out[i] = f2bf(in[i]);
}

// ---------------------------------------------------------------------------
// Fused QKV projection: y = x @ W^T + b, scattered to [B,H,S,DH] bf16.
// Block: 256 thr (8 waves). Tile: 128(M) x 128(N), K-step 32.
// grid = (M/128, N/128, 3)   z selects Q/K/V
// ---------------------------------------------------------------------------
__launch_bounds__(256) __global__
void gemm_qkv(const unsigned short* __restrict__ xb,
              const unsigned short* __restrict__ wq,
              const unsigned short* __restrict__ wk,
              const unsigned short* __restrict__ wv,
              const float* __restrict__ bqp, const float* __restrict__ bkp,
              const float* __restrict__ bvp,
              unsigned short* __restrict__ Qb, unsigned short* __restrict__ Kb,
              unsigned short* __restrict__ Vb) {
  const unsigned short* W = blockIdx.z == 0 ? wq : (blockIdx.z == 1 ? wk : wv);
  const float* bias = blockIdx.z == 0 ? bqp : (blockIdx.z == 1 ? bkp : bvp);
  unsigned short* Out = blockIdx.z == 0 ? Qb : (blockIdx.z == 1 ? Kb : Vb);

  __shared__ unsigned short sB[128 * 40];  // 128 N-rows x 32 K (+8 pad)

  const int tid = threadIdx.x;
  const int wave = tid >> 5, lane = tid & 31;
  const int hi = (lane >> 4) & 1;
  const int ln = lane & 15;

  const int m0 = blockIdx.x * 128 + wave * 16;
  const int n0 = blockIdx.y * 128;

  v8f zero = {};
  v8f acc[8];
#pragma unroll
  for (int j = 0; j < 8; ++j) acc[j] = zero;

  const unsigned short* arow = xb + (size_t)(m0 + ln) * D_;

  for (int k0 = 0; k0 < D_; k0 += 32) {
    __syncthreads();
#if HAVE_TDM
    if (wave == 0) {
      // 128 rows x 32 bf16, row stride D_; LDS rows padded 32 -> 40 bf16
      tdm_load_2d((unsigned)(uintptr_t)(&sB[0]),
                  W + (size_t)n0 * D_ + k0,
                  /*tensor_d0=*/D_, /*tensor_d1=*/4096,
                  /*tile_d0=*/32, /*tile_d1=*/128, /*stride0=*/D_,
                  /*pad_interval_log: 16 dwords=*/3, /*pad 4 dwords=*/3);
    }
#else
    {
      int fn = tid & 127, fk = (tid >> 7) << 4;
      const unsigned short* src = W + (size_t)(n0 + fn) * D_ + k0 + fk;
      u32x4 a0 = *(const u32x4*)(src);
      u32x4 a1 = *(const u32x4*)(src + 8);
      *(u32x4*)(&sB[fn * 40 + fk]) = a0;
      *(u32x4*)(&sB[fn * 40 + fk + 8]) = a1;
    }
#endif
    // overlap: A fragment from global while the tile DMA is in flight
    FragB af;
    const unsigned short* ap = arow + k0 + (hi ? 8 : 0);
    af.q[0] = *(const u32x4*)(ap);
    af.q[1] = *(const u32x4*)(ap + 16);
#if HAVE_TDM
    if (wave == 0) __builtin_amdgcn_s_wait_tensorcnt((unsigned short)0);
#endif
    __syncthreads();

    FragB bfr[8];  // preload all B fragments: batch ds_loads, batch waits
#pragma unroll
    for (int j = 0; j < 8; ++j) {
      const unsigned short* bp = &sB[(j * 16 + ln) * 40 + (hi ? 8 : 0)];
      bfr[j].q[0] = *(const u32x4*)(bp);
      bfr[j].q[1] = *(const u32x4*)(bp + 16);
    }
#pragma unroll
    for (int j = 0; j < 8; ++j)
      acc[j] = __builtin_amdgcn_wmma_f32_16x16x32_bf16(
          false, af.f, false, bfr[j].f, (short)0, acc[j], false, false);
  }

#pragma unroll
  for (int j = 0; j < 8; ++j) {
    int n = n0 + j * 16 + ln;
    float bv = bias[n];
    int h = n >> 6, dh = n & 63;
#pragma unroll
    for (int r = 0; r < 8; ++r) {
      int m = m0 + r + hi * 8;
      int b = m >> 11, s = m & (S_ - 1);
      Out[(((size_t)(b * H_ + h) * S_) + s) * DH_ + dh] = f2bf(acc[j][r] + bv);
    }
  }
}

// ---------------------------------------------------------------------------
// Flash attention: per (b,h), 128 query rows per block, key blocks of 128.
// grid = (S/128, B*H), 256 thr (8 waves; each wave owns 16 query rows).
// ---------------------------------------------------------------------------
__launch_bounds__(256) __global__
void attn(const unsigned short* __restrict__ Qb,
          const unsigned short* __restrict__ Kb,
          const unsigned short* __restrict__ Vb,
          const int* __restrict__ mask, unsigned short* __restrict__ Ob) {
  // sKP doubles as K tile [128][72] (scores phase) and P buffers
  // [8 waves][16][136] (PV phase), separated by a barrier.
  __shared__ unsigned short sKP[8 * 16 * 136];  // 34,816 B
  __shared__ unsigned short sVT[64 * 136];      // 17,408 B  (V transposed)

  const int tid = threadIdx.x;
  const int wave = tid >> 5, lane = tid & 31;
  const int hi = (lane >> 4) & 1;
  const int ln = lane & 15;
  const int bh = blockIdx.y;
  const int b = bh >> 4, h = bh & 15;
  const int qw = blockIdx.x * 128 + wave * 16;

  const size_t head = (size_t)(b * H_ + h) * S_ * DH_;
  const unsigned short* qbase = Qb + head;
  const unsigned short* kbase = Kb + head;
  const unsigned short* vbase = Vb + head;

  // resident Q fragments (16 x 64 = two 16x32 A fragments)
  FragB aq[2];
  {
    const unsigned short* qp = qbase + (size_t)(qw + ln) * DH_ + (hi ? 8 : 0);
    aq[0].q[0] = *(const u32x4*)(qp);
    aq[0].q[1] = *(const u32x4*)(qp + 16);
    aq[1].q[0] = *(const u32x4*)(qp + 32);
    aq[1].q[1] = *(const u32x4*)(qp + 48);
  }

  v8f zero = {};
  float mrow[8], lrow[8];
  v8f o[4];
#pragma unroll
  for (int r = 0; r < 8; ++r) { mrow[r] = -3.0e38f; lrow[r] = 0.f; }
#pragma unroll
  for (int t = 0; t < 4; ++t) o[t] = zero;

  unsigned short* pbuf = &sKP[wave * 16 * 136];
  const int frow = tid & 127;        // tile fill: key row
  const int fdb = (tid >> 7) << 5;   // tile fill: dh base (0 or 32)

  for (int kb = 0; kb < S_; kb += 128) {
    __syncthreads();  // prev-iter P / VT consumers done
#if HAVE_TDM
    if (wave == 0) {
      // K tile: 128 rows x 64 bf16, stride DH_; LDS rows padded 64 -> 72 bf16
      tdm_load_2d((unsigned)(uintptr_t)(&sKP[0]),
                  kbase + (size_t)kb * DH_,
                  /*tensor_d0=*/DH_, /*tensor_d1=*/S_,
                  /*tile_d0=*/DH_, /*tile_d1=*/128, /*stride0=*/DH_,
                  /*pad_interval_log: 32 dwords=*/4, /*pad 4 dwords=*/3);
    }
#endif
    {
      const unsigned short* vs = vbase + (size_t)(kb + frow) * DH_ + fdb;
      if (kb + 128 < S_) __builtin_prefetch(vs + 128 * DH_, 0, 0);
#if !HAVE_TDM
      const unsigned short* ks = kbase + (size_t)(kb + frow) * DH_ + fdb;
#pragma unroll
      for (int i = 0; i < 4; ++i) {
        u32x4 v = *(const u32x4*)(ks + i * 8);
        *(u32x4*)(&sKP[frow * 72 + fdb + i * 8]) = v;
      }
#endif
#pragma unroll
      for (int i = 0; i < 4; ++i) {  // transpose V into sVT[dh][row]
        Pack8 p;
        p.v = *(const u32x4*)(vs + i * 8);
#pragma unroll
        for (int d = 0; d < 8; ++d) sVT[(fdb + i * 8 + d) * 136 + frow] = p.e[d];
      }
    }
#if HAVE_TDM
    if (wave == 0) __builtin_amdgcn_s_wait_tensorcnt((unsigned short)0);
#endif
    __syncthreads();

    // ---- scores: S = Q @ K^T  (8 tiles of 16x16, K-dim 64 = 2 WMMAs) ----
    v8f sc[8];
#pragma unroll
    for (int j = 0; j < 8; ++j) {
      sc[j] = zero;
      const unsigned short* bp = &sKP[(j * 16 + ln) * 72 + (hi ? 8 : 0)];
      FragB b0, b1;
      b0.q[0] = *(const u32x4*)(bp);
      b0.q[1] = *(const u32x4*)(bp + 16);
      b1.q[0] = *(const u32x4*)(bp + 32);
      b1.q[1] = *(const u32x4*)(bp + 48);
      sc[j] = __builtin_amdgcn_wmma_f32_16x16x32_bf16(
          false, aq[0].f, false, b0.f, (short)0, sc[j], false, false);
      sc[j] = __builtin_amdgcn_wmma_f32_16x16x32_bf16(
          false, aq[1].f, false, b1.f, (short)0, sc[j], false, false);
    }

    __syncthreads();  // all waves done reading K tile; sKP becomes P buffer

    // ---- mask + online softmax (row stats per lane, reduce over 16 lanes) --
#pragma unroll
    for (int r = 0; r < 8; ++r) {
      int qrow = qw + r + hi * 8;
      const int* mp = mask + ((size_t)b * S_ + qrow) * S_ + kb + ln;
      float mx = -3.0e38f;
#pragma unroll
      for (int j = 0; j < 8; ++j) {
        float s = sc[j][r] * 0.125f;  // 1/sqrt(64)
        if (mp[j * 16] == 0) s = NEGV;
        sc[j][r] = s;
        mx = fmaxf(mx, s);
      }
#pragma unroll
      for (int dlt = 1; dlt < 16; dlt <<= 1)
        mx = fmaxf(mx, __shfl_xor(mx, dlt, 32));
      float mnew = fmaxf(mrow[r], mx);
      float corr = __expf(mrow[r] - mnew);
      mrow[r] = mnew;
      float rs = 0.f;
#pragma unroll
      for (int j = 0; j < 8; ++j) {
        float p = __expf(sc[j][r] - mnew);
        rs += p;
        pbuf[(r + hi * 8) * 136 + j * 16 + ln] = f2bf(p);
      }
#pragma unroll
      for (int dlt = 1; dlt < 16; dlt <<= 1) rs += __shfl_xor(rs, dlt, 32);
      lrow[r] = lrow[r] * corr + rs;
#pragma unroll
      for (int t = 0; t < 4; ++t) o[t][r] *= corr;
    }

    // ---- O += P @ V  (K-dim 128 = 4 chunks of 32; 4 dh tiles) ----
#pragma unroll
    for (int c = 0; c < 4; ++c) {
      FragB ap;
      const unsigned short* pp = &pbuf[ln * 136 + c * 32 + (hi ? 8 : 0)];
      ap.q[0] = *(const u32x4*)(pp);
      ap.q[1] = *(const u32x4*)(pp + 16);
      FragB bvf[4];  // preload all 4 V fragments for this K-chunk
#pragma unroll
      for (int t = 0; t < 4; ++t) {
        const unsigned short* vp =
            &sVT[(t * 16 + ln) * 136 + c * 32 + (hi ? 8 : 0)];
        bvf[t].q[0] = *(const u32x4*)(vp);
        bvf[t].q[1] = *(const u32x4*)(vp + 16);
      }
#pragma unroll
      for (int t = 0; t < 4; ++t)
        o[t] = __builtin_amdgcn_wmma_f32_16x16x32_bf16(
            false, ap.f, false, bvf[t].f, (short)0, o[t], false, false);
    }
  }

  // normalize and scatter to [B*S, D] bf16 (merge-heads layout)
#pragma unroll
  for (int t = 0; t < 4; ++t) {
#pragma unroll
    for (int r = 0; r < 8; ++r) {
      int qrow = qw + r + hi * 8;
      float y = o[t][r] / lrow[r];
      Ob[(size_t)(b * S_ + qrow) * D_ + h * DH_ + t * 16 + ln] = f2bf(y);
    }
  }
}

// ---------------------------------------------------------------------------
// Output projection: out = Ob @ Wo^T + bo (fp32 result)
// ---------------------------------------------------------------------------
__launch_bounds__(256) __global__
void gemm_out(const unsigned short* __restrict__ Ab,
              const unsigned short* __restrict__ W,
              const float* __restrict__ bias, float* __restrict__ out) {
  __shared__ unsigned short sB[128 * 40];

  const int tid = threadIdx.x;
  const int wave = tid >> 5, lane = tid & 31;
  const int hi = (lane >> 4) & 1;
  const int ln = lane & 15;

  const int m0 = blockIdx.x * 128 + wave * 16;
  const int n0 = blockIdx.y * 128;

  v8f zero = {};
  v8f acc[8];
#pragma unroll
  for (int j = 0; j < 8; ++j) acc[j] = zero;

  const unsigned short* arow = Ab + (size_t)(m0 + ln) * D_;

  for (int k0 = 0; k0 < D_; k0 += 32) {
    __syncthreads();
#if HAVE_TDM
    if (wave == 0) {
      tdm_load_2d((unsigned)(uintptr_t)(&sB[0]),
                  W + (size_t)n0 * D_ + k0,
                  /*tensor_d0=*/D_, /*tensor_d1=*/4096,
                  /*tile_d0=*/32, /*tile_d1=*/128, /*stride0=*/D_,
                  /*pad_interval_log=*/3, /*pad_amount_dw=*/3);
    }
#else
    {
      int fn = tid & 127, fk = (tid >> 7) << 4;
      const unsigned short* src = W + (size_t)(n0 + fn) * D_ + k0 + fk;
      u32x4 a0 = *(const u32x4*)(src);
      u32x4 a1 = *(const u32x4*)(src + 8);
      *(u32x4*)(&sB[fn * 40 + fk]) = a0;
      *(u32x4*)(&sB[fn * 40 + fk + 8]) = a1;
    }
#endif
    FragB af;
    const unsigned short* ap = arow + k0 + (hi ? 8 : 0);
    af.q[0] = *(const u32x4*)(ap);
    af.q[1] = *(const u32x4*)(ap + 16);
#if HAVE_TDM
    if (wave == 0) __builtin_amdgcn_s_wait_tensorcnt((unsigned short)0);
#endif
    __syncthreads();

    FragB bfr[8];
#pragma unroll
    for (int j = 0; j < 8; ++j) {
      const unsigned short* bp = &sB[(j * 16 + ln) * 40 + (hi ? 8 : 0)];
      bfr[j].q[0] = *(const u32x4*)(bp);
      bfr[j].q[1] = *(const u32x4*)(bp + 16);
    }
#pragma unroll
    for (int j = 0; j < 8; ++j)
      acc[j] = __builtin_amdgcn_wmma_f32_16x16x32_bf16(
          false, af.f, false, bfr[j].f, (short)0, acc[j], false, false);
  }

#pragma unroll
  for (int j = 0; j < 8; ++j) {
    int n = n0 + j * 16 + ln;
    float bv = bias[n];
#pragma unroll
    for (int r = 0; r < 8; ++r) {
      int m = m0 + r + hi * 8;
      out[(size_t)m * D_ + n] = acc[j][r] + bv;
    }
  }
}

// ---------------------------------------------------------------------------
extern "C" void kernel_launch(void* const* d_in, const int* in_sizes, int n_in,
                              void* d_out, int out_size, void* d_ws,
                              size_t ws_size, hipStream_t stream) {
  (void)in_sizes; (void)n_in; (void)out_size; (void)ws_size;

  const float* x = (const float*)d_in[0];
  const int* mask = (const int*)d_in[1];
  const float* Wq = (const float*)d_in[2];
  const float* bq = (const float*)d_in[3];
  const float* Wk = (const float*)d_in[4];
  const float* bk = (const float*)d_in[5];
  const float* Wv = (const float*)d_in[6];
  const float* bv = (const float*)d_in[7];
  const float* Wo = (const float*)d_in[8];
  const float* bo = (const float*)d_in[9];
  float* out = (float*)d_out;

  char* ws = (char*)d_ws;
  unsigned short* xb = (unsigned short*)(ws);                       // 8 MB
  unsigned short* wqb = (unsigned short*)(ws + (8ull << 20));       // 2 MB
  unsigned short* wkb = wqb + (1 << 20);
  unsigned short* wvb = wkb + (1 << 20);
  unsigned short* wob = wvb + (1 << 20);
  unsigned short* Qb = (unsigned short*)(ws + (16ull << 20));       // 8 MB
  unsigned short* Kb = (unsigned short*)(ws + (24ull << 20));       // 8 MB
  unsigned short* Vb = (unsigned short*)(ws + (32ull << 20));       // 8 MB
  unsigned short* Ob = (unsigned short*)(ws + (40ull << 20));       // 8 MB

  const int nx = B_ * S_ * D_;   // 4,194,304
  const int nw = D_ * D_;        // 1,048,576
  cvt_f32_to_bf16<<<(nx + 255) / 256, 256, 0, stream>>>(x, xb, nx);
  cvt_f32_to_bf16<<<(nw + 255) / 256, 256, 0, stream>>>(Wq, wqb, nw);
  cvt_f32_to_bf16<<<(nw + 255) / 256, 256, 0, stream>>>(Wk, wkb, nw);
  cvt_f32_to_bf16<<<(nw + 255) / 256, 256, 0, stream>>>(Wv, wvb, nw);
  cvt_f32_to_bf16<<<(nw + 255) / 256, 256, 0, stream>>>(Wo, wob, nw);

  gemm_qkv<<<dim3((B_ * S_) / 128, D_ / 128, 3), 256, 0, stream>>>(
      xb, wqb, wkb, wvb, bq, bk, bv, Qb, Kb, Vb);

  attn<<<dim3(S_ / 128, B_ * H_), 256, 0, stream>>>(Qb, Kb, Vb, mask, Ob);

  gemm_out<<<dim3((B_ * S_) / 128, D_ / 128), 256, 0, stream>>>(Ob, wob, bo, out);
}
